// VersorAttention_62113817034793
// MI455X (gfx1250) — matrix operand order repaired and checked by reading, compiled, not run on hardware
//
#include <hip/hip_runtime.h>
#include <hip/hip_bf16.h>

typedef unsigned short u16;
typedef __attribute__((ext_vector_type(16))) __bf16 v16bf;
typedef __attribute__((ext_vector_type(8)))  __bf16 v8bf;
typedef __attribute__((ext_vector_type(8)))  float  v8f;
typedef __attribute__((ext_vector_type(8)))  unsigned v8u;
typedef int v4i __attribute__((vector_size(4 * sizeof(int))));
typedef __attribute__((address_space(1))) v4i* gv4i_p;   // global
typedef __attribute__((address_space(3))) v4i* lv4i_p;   // LDS

#define GA     32
#define EDIM   256
#define SEQ    512
#define NH     8
#define HD     32
#define BHCNT  16
#define BSTOT  1024          // B*S
#define NGEO   32768         // BSTOT*GA
#define LDSTR  72            // LDS tile row stride (u16), 144B -> 16B aligned, conflict-breaking

// ---------------- helpers ----------------

__device__ __forceinline__ u16 f2bf(float f) {
  unsigned u = __float_as_uint(f);
  unsigned r = u + 0x7FFFu + ((u >> 16) & 1u);
  return (u16)(r >> 16);
}

// Cayley sign for e_a * e_b in Cl(4,1): returns 1 if negative
__device__ __forceinline__ int geo_sign_neg(int a, int b) {
  int ss = 0;
  int aa = a >> 1;
  while (aa) { ss += __popc(aa & b); aa >>= 1; }
  ss ^= ((a & b) >> 4) & 1;     // metric: e4^2 = -1
  return ss & 1;
}

__device__ __forceinline__ v16bf cat8(v8bf lo, v8bf hi) {
  return __builtin_shufflevector(lo, hi, 0,1,2,3,4,5,6,7,8,9,10,11,12,13,14,15);
}

// async global -> LDS copy of 16B per lane (CDNA5 GLOBAL_LOAD_ASYNC_TO_LDS_B128, ASYNCcnt)
__device__ __forceinline__ void async_copy16(const u16* g, u16* l) {
#if __has_builtin(__builtin_amdgcn_global_load_async_to_lds_b128)
  __builtin_amdgcn_global_load_async_to_lds_b128(
      (gv4i_p)(v4i*)g, (lv4i_p)(v4i*)l, 0, 0);
#else
  unsigned loff = (unsigned)(unsigned long long)(__attribute__((address_space(3))) const char*)l;
  asm volatile("global_load_async_to_lds_b128 %0, %1, off" :: "v"(loff), "v"(g) : "memory");
#endif
}

template<int N>
__device__ __forceinline__ void wait_async() {
#if __has_builtin(__builtin_amdgcn_s_wait_asynccnt)
  __builtin_amdgcn_s_wait_asynccnt(N);
#else
  asm volatile("s_wait_asynccnt %0" :: "n"(N) : "memory");
#endif
}

// A fragment 16x32 bf16 from 64-deep tile: lane L<16 -> row L, K 0..7 & 16..23 ; L>=16 -> K 8..15 & 24..31
__device__ __forceinline__ v16bf load_fragA(const u16* lds, int row_base, int lane, int kc) {
  int row  = row_base + (lane & 15);
  int koff = (lane >> 4) << 3;                   // 0 or 8
  const u16* p = lds + row * LDSTR + kc * 32 + koff;
  v8bf lo = *reinterpret_cast<const v8bf*>(p);
  v8bf hi = *reinterpret_cast<const v8bf*>(p + 16);
  return cat8(lo, hi);
}

// B fragment 32x16 bf16 (LDS holds [n][k]): lane L<16 -> col L, K 0..7 & 16..23 ; L>=16 -> col L-16, K 8..15 & 24..31
__device__ __forceinline__ v16bf load_fragB(const u16* lds, int row_base, int lane, int kc) {
  int row   = row_base + (lane & 15);
  int kbase = (lane >> 4) << 3;                  // 0 or 8
  const u16* p = lds + row * LDSTR + kc * 32 + kbase;
  v8bf lo = *reinterpret_cast<const v8bf*>(p);
  v8bf hi = *reinterpret_cast<const v8bf*>(p + 16);
  return cat8(lo, hi);
}

__device__ __forceinline__ v16bf xor_frag(v16bf f, unsigned m) {
  union { v16bf h; v8u u; } t;
  t.h = f;
  t.u ^= m;
  return t.h;
}

__device__ __forceinline__ v8f wmma_bf16(v16bf a, v16bf b, v8f c) {
  return __builtin_amdgcn_wmma_f32_16x16x32_bf16(false, a, false, b, (short)0, c, false, false);
}

// ---------------- prep kernels ----------------

// x[bs][i][g] f32  ->  xbT[bs][g][i] bf16
__global__ __launch_bounds__(256)
void prep_x_kernel(const float* __restrict__ x, u16* __restrict__ xbT) {
  int t = blockIdx.x * 256 + threadIdx.x;        // 8388608
  int g = t & 31, i = (t >> 5) & 255, bs = t >> 13;
  xbT[((size_t)bs * GA + g) * EDIM + i] = f2bf(x[t]);
}

// w[o][i][m] f32 (x4)  ->  wbT[w][m][o][i] bf16
__global__ __launch_bounds__(256)
void prep_w_kernel(const float* __restrict__ w0, const float* __restrict__ w1,
                   const float* __restrict__ w2, const float* __restrict__ w3,
                   u16* __restrict__ wbT) {
  int t = blockIdx.x * 256 + threadIdx.x;        // 4 * 2097152
  int w = t >> 21; int rem = t & 2097151;
  int m = rem & 31, i = (rem >> 5) & 255, o = rem >> 13;
  const float* src = (w == 0) ? w0 : (w == 1) ? w1 : (w == 2) ? w2 : w3;
  wbT[(((size_t)w * GA + m) * EDIM + o) * EDIM + i] = f2bf(src[rem]);
}

// ---------------- geo linear: C[o, (bs,l)] = sum_m W_m x ----------------
// Wb: [32][256][256] bf16, XbT: [1024][32][256] bf16, Out: [1024][256][32] f32
// Async double-buffered staging, K-stage depth 64, 8 waves, 128x128 C tile.
__global__ __launch_bounds__(256)
void geo_gemm_kernel(const u16* __restrict__ Wb, const u16* __restrict__ XbT,
                     float* __restrict__ Out) {
  __shared__ __align__(16) u16 lA[2][128 * LDSTR];
  __shared__ __align__(16) u16 lB[2][128 * LDSTR];
  int tid = threadIdx.x, lane = tid & 31, wave = tid >> 5;
  int wm = wave & 1, wn = wave >> 1;
  int n0 = blockIdx.x * 128;
  int o0 = blockIdx.y * 128;

  v8f acc[4][2];
  v8f zero = {0.f,0.f,0.f,0.f,0.f,0.f,0.f,0.f};
#pragma unroll
  for (int a = 0; a < 4; ++a)
#pragma unroll
    for (int b = 0; b < 2; ++b) acc[a][b] = zero;

  int sr = tid >> 1, sh = tid & 1;               // staging row / 32-elem half
  int ncol = n0 + sr;
  int bs = ncol >> 5, l = ncol & 31;
  // column blade-components owned by this lane's two B fragments
  int lcol0 = lane & 15;
  int lcol1 = 16 + (lane & 15);

  const int NSTEP = GA * (EDIM / 64);            // 128 stages of K=64

  auto issue = [&](int step, int buf) {
    int m = step >> 2, kt = step & 3;
    const u16* ga = Wb  + ((size_t)m * EDIM + (o0 + sr)) * EDIM + kt * 64 + sh * 32;
    const u16* gb = XbT + ((size_t)bs * GA + (l ^ m)) * EDIM + kt * 64 + sh * 32;
    u16* la = &lA[buf][sr * LDSTR + sh * 32];
    u16* lb = &lB[buf][sr * LDSTR + sh * 32];
#pragma unroll
    for (int c = 0; c < 4; ++c) {
      async_copy16(ga + c * 8, la + c * 8);
      async_copy16(gb + c * 8, lb + c * 8);
    }
  };

  issue(0, 0);
#pragma unroll 1
  for (int step = 0; step < NSTEP; ++step) {
    if (step + 1 < NSTEP) { issue(step + 1, (step + 1) & 1); wait_async<8>(); }
    else                  { wait_async<0>(); }
    __syncthreads();
    int m = step >> 2;
    unsigned bm0 = geo_sign_neg(m, lcol0 ^ m) ? 0x80008000u : 0u;
    unsigned bm1 = geo_sign_neg(m, lcol1 ^ m) ? 0x80008000u : 0u;
    const u16* cA = lA[step & 1];
    const u16* cB = lB[step & 1];
#pragma unroll
    for (int kc = 0; kc < 2; ++kc) {
      v16bf afr[4], bfr[2];
#pragma unroll
      for (int tm = 0; tm < 4; ++tm) afr[tm] = load_fragA(cA, wm * 64 + tm * 16, lane, kc);
      bfr[0] = xor_frag(load_fragB(cB, wn * 32 +  0, lane, kc), bm0);
      bfr[1] = xor_frag(load_fragB(cB, wn * 32 + 16, lane, kc), bm1);
#pragma unroll
      for (int tm = 0; tm < 4; ++tm)
#pragma unroll
        for (int tn = 0; tn < 2; ++tn)
          acc[tm][tn] = wmma_bf16(afr[tm], bfr[tn], acc[tm][tn]);
    }
    __syncthreads();
  }

  int colL = lane & 15;
  int rtop = (lane >> 4) << 3;
#pragma unroll
  for (int tm = 0; tm < 4; ++tm)
#pragma unroll
    for (int tn = 0; tn < 2; ++tn) {
      int nc = n0 + wn * 32 + tn * 16 + colL;
      int bs2 = nc >> 5, l2 = nc & 31;
      int ob = o0 + wm * 64 + tm * 16 + rtop;
      float* op = Out + (size_t)bs2 * (EDIM * GA) + l2;
#pragma unroll
      for (int r = 0; r < 8; ++r) op[(size_t)(ob + r) * GA] = acc[tm][tn][r];
    }
}

// ---------------- versor norm + pack into attention layouts ----------------
// G: [1024][256][32] f32. WHICH: 0 -> qsig [16][512][1024]+qn2, 1 -> kbf+kn2, 2 -> vT [16][1024][512]
template<int WHICH>
__global__ __launch_bounds__(256)
void norm_pack_kernel(const float* __restrict__ G, u16* __restrict__ pk,
                      float* __restrict__ n2) {
  __shared__ float part[256];
  int bs = blockIdx.x, o = threadIdx.x;
  const float* p = G + ((size_t)bs * EDIM + o) * GA;
  float v[GA]; float s = 0.f;
#pragma unroll
  for (int l = 0; l < GA; ++l) { v[l] = p[l]; s += v[l] * v[l]; }
  float inv = rsqrtf(s + 1e-6f);
  int b = bs >> 9, sidx = bs & 511, h = o >> 5, d = o & 31, bh = b * NH + h;
  if (WHICH == 2) {
    size_t base = ((size_t)bh * 1024 + d * GA) * SEQ + sidx;
#pragma unroll
    for (int l = 0; l < GA; ++l) pk[base + (size_t)l * SEQ] = f2bf(v[l] * inv);
  } else {
    size_t base = ((size_t)(bh * SEQ + sidx)) * 1024 + d * GA;
#pragma unroll
    for (int l = 0; l < GA; ++l) {
      float xv = v[l] * inv;
      if (WHICH == 0 && geo_sign_neg(l, l)) xv = -xv;   // metric diagonal
      pk[base + l] = f2bf(xv);
    }
    part[o] = s * inv * inv;                            // sum_l qhat^2 for this o
    __syncthreads();
    if (o < NH) {
      float a2 = 0.f;
#pragma unroll
      for (int d2 = 0; d2 < 32; ++d2) a2 += part[o * 32 + d2];
      n2[(b * NH + o) * SEQ + sidx] = a2;
    }
  }
}

// ---------------- batched plain WMMA GEMM, M=512 ----------------
// A: [16][512][KDIM] bf16, BT: [16][NDIM][KDIM] bf16
// MODE 0: C=acc (f32); MODE 1: C = C + 0.1*acc, mask; MODE 2: scatter bf16 -> aoT
template<int MODE, int KDIM, int NDIM>
__global__ __launch_bounds__(256)
void gemm_kernel(const u16* __restrict__ A, const u16* __restrict__ BT,
                 float* __restrict__ C, u16* __restrict__ aoT,
                 const int* __restrict__ maskp) {
  __shared__ __align__(16) u16 lA[2][128 * LDSTR];
  __shared__ __align__(16) u16 lB[2][128 * LDSTR];
  int tid = threadIdx.x, lane = tid & 31, wave = tid >> 5;
  int wm = wave & 1, wn = wave >> 1;
  int bh = blockIdx.z;
  int n0 = blockIdx.x * 128, m0 = blockIdx.y * 128;

  v8f acc[4][2];
  v8f zero = {0.f,0.f,0.f,0.f,0.f,0.f,0.f,0.f};
#pragma unroll
  for (int a = 0; a < 4; ++a)
#pragma unroll
    for (int b = 0; b < 2; ++b) acc[a][b] = zero;

  int sr = tid >> 1, sh = tid & 1;
  const u16* arow = A  + ((size_t)bh * SEQ  + (m0 + sr)) * KDIM + sh * 32;
  const u16* brow = BT + ((size_t)bh * NDIM + (n0 + sr)) * KDIM + sh * 32;

  const int NSTEP = KDIM / 64;

  auto issue = [&](int step, int buf) {
    const u16* ga = arow + step * 64;
    const u16* gb = brow + step * 64;
    u16* la = &lA[buf][sr * LDSTR + sh * 32];
    u16* lb = &lB[buf][sr * LDSTR + sh * 32];
#pragma unroll
    for (int c = 0; c < 4; ++c) {
      async_copy16(ga + c * 8, la + c * 8);
      async_copy16(gb + c * 8, lb + c * 8);
    }
  };

  issue(0, 0);
#pragma unroll 1
  for (int step = 0; step < NSTEP; ++step) {
    if (step + 1 < NSTEP) { issue(step + 1, (step + 1) & 1); wait_async<8>(); }
    else                  { wait_async<0>(); }
    __syncthreads();
    const u16* cA = lA[step & 1];
    const u16* cB = lB[step & 1];
#pragma unroll
    for (int kc = 0; kc < 2; ++kc) {
      v16bf afr[4], bfr[2];
#pragma unroll
      for (int tm = 0; tm < 4; ++tm) afr[tm] = load_fragA(cA, wm * 64 + tm * 16, lane, kc);
#pragma unroll
      for (int tn = 0; tn < 2; ++tn) bfr[tn] = load_fragB(cB, wn * 32 + tn * 16, lane, kc);
#pragma unroll
      for (int tm = 0; tm < 4; ++tm)
#pragma unroll
        for (int tn = 0; tn < 2; ++tn)
          acc[tm][tn] = wmma_bf16(afr[tm], bfr[tn], acc[tm][tn]);
    }
    __syncthreads();
  }

  int colL = lane & 15;
  int rtop = (lane >> 4) << 3;
#pragma unroll
  for (int tm = 0; tm < 4; ++tm)
#pragma unroll
    for (int tn = 0; tn < 2; ++tn) {
      int jg = n0 + wn * 32 + tn * 16 + colL;
      int itop = m0 + wm * 64 + tm * 16 + rtop;
#pragma unroll
      for (int r = 0; r < 8; ++r) {
        int ig = itop + r;
        float av = acc[tm][tn][r];
        if (MODE == 0) {
          C[((size_t)bh * SEQ + ig) * NDIM + jg] = av;
        } else if (MODE == 1) {
          size_t idx = ((size_t)bh * SEQ + ig) * NDIM + jg;
          float val = C[idx] + 0.1f * av;
          int b = bh >> 3;
          if (maskp[b * SEQ + jg] == 0) val = -1e9f;
          C[idx] = val;
        } else {                                   // MODE 2: attn @ v -> aoT[bs][l][o]
          int b = bh >> 3, h = bh & 7;
          int bs2 = b * SEQ + ig;
          int d = jg >> 5, l2 = jg & 31;
          int o = h * HD + d;
          aoT[((size_t)bs2 * GA + l2) * EDIM + o] = f2bf(av);
        }
      }
    }
}

// ---------------- score fix: bivector term + transpose to bf16 ----------------
__global__ __launch_bounds__(256)
void score_fix_kernel(float* __restrict__ sf, const float* __restrict__ qn2,
                      const float* __restrict__ kn2, const float* __restrict__ gamma,
                      u16* __restrict__ scoreT) {
  int t = blockIdx.x * 256 + threadIdx.x;        // 16*512*512
  int j = t & 511, i = (t >> 9) & 511, bh = t >> 18;
  float ss = sf[t];
  float qn = qn2[bh * SEQ + i], kn = kn2[bh * SEQ + j];
  float bn = sqrtf(fmaxf(qn * kn - ss * ss, 0.f) + 1e-6f);
  float sc = (ss + gamma[0] * bn) * 0.17677669529663687f;   // 1/sqrt(HD)
  sf[t] = sc;
  scoreT[((size_t)bh << 18) + ((size_t)j << 9) + i] = f2bf(sc);
}

// ---------------- row softmax f32[.,512] -> bf16[.,512] ----------------
__global__ __launch_bounds__(256)
void softmax_kernel(const float* __restrict__ in, u16* __restrict__ out) {
  __shared__ float red[256];
  int row = blockIdx.x;
  const float* rp = in + (size_t)row * SEQ;
  int t = threadIdx.x;
  float a = rp[t], b = rp[t + 256];
  red[t] = fmaxf(a, b);
  __syncthreads();
  for (int s2 = 128; s2 > 0; s2 >>= 1) {
    if (t < s2) red[t] = fmaxf(red[t], red[t + s2]);
    __syncthreads();
  }
  float m = red[0];
  __syncthreads();
  float e0 = __expf(a - m), e1 = __expf(b - m);
  red[t] = e0 + e1;
  __syncthreads();
  for (int s2 = 128; s2 > 0; s2 >>= 1) {
    if (t < s2) red[t] += red[t + s2];
    __syncthreads();
  }
  float inv = 1.f / red[0];
  out[(size_t)row * SEQ + t]       = f2bf(e0 * inv);
  out[(size_t)row * SEQ + t + 256] = f2bf(e1 * inv);
}

// ---------------- final in-place versor norm on d_out ----------------
__global__ __launch_bounds__(256)
void final_norm_kernel(float* __restrict__ out) {
  int t = blockIdx.x * 256 + threadIdx.x;        // 262144 = BSTOT*EDIM
  float* p = out + (size_t)t * GA;
  float v[GA]; float s = 0.f;
#pragma unroll
  for (int l = 0; l < GA; ++l) { v[l] = p[l]; s += v[l] * v[l]; }
  float inv = rsqrtf(s + 1e-6f);
#pragma unroll
  for (int l = 0; l < GA; ++l) p[l] = v[l] * inv;
}

// ---------------- host launcher ----------------

extern "C" void kernel_launch(void* const* d_in, const int* in_sizes, int n_in,
                              void* d_out, int out_size, void* d_ws, size_t ws_size,
                              hipStream_t stream) {
  (void)in_sizes; (void)n_in; (void)out_size; (void)ws_size;
  const float* x     = (const float*)d_in[0];
  const float* wq    = (const float*)d_in[1];
  const float* wk    = (const float*)d_in[2];
  const float* wv    = (const float*)d_in[3];
  const float* wo    = (const float*)d_in[4];
  const float* gamma = (const float*)d_in[5];
  const int*   maskp = (const int*)d_in[6];
  float* outp = (float*)d_out;

  char* ws = (char*)d_ws;
  size_t off = 0;
  auto alloc = [&](size_t bytes) { size_t o = off; off += (bytes + 255) & ~(size_t)255; return o; };
  u16*   xbT    = (u16*)  (ws + alloc((size_t)BSTOT * GA * EDIM * 2));        // 16 MB
  u16*   wbT    = (u16*)  (ws + alloc((size_t)4 * GA * EDIM * EDIM * 2));     // 16 MB
  float* gbuf   = (float*)(ws + alloc((size_t)BSTOT * EDIM * GA * 4));        // 32 MB
  u16*   qsig   = (u16*)  (ws + alloc((size_t)BHCNT * SEQ * 1024 * 2));       // 16 MB
  u16*   kbf    = (u16*)  (ws + alloc((size_t)BHCNT * SEQ * 1024 * 2));       // 16 MB
  u16*   vT     = (u16*)  (ws + alloc((size_t)BHCNT * 1024 * SEQ * 2));       // 16 MB
  float* qn2    = (float*)(ws + alloc((size_t)BHCNT * SEQ * 4));
  float* kn2    = (float*)(ws + alloc((size_t)BHCNT * SEQ * 4));
  float* sf32   = (float*)(ws + alloc((size_t)BHCNT * SEQ * SEQ * 4));        // 16 MB
  u16*   scoreT = (u16*)  (ws + alloc((size_t)BHCNT * SEQ * SEQ * 2));        // 8 MB
  u16*   Pbuf   = (u16*)  (ws + alloc((size_t)BHCNT * SEQ * SEQ * 2));        // 8 MB (P, then attn)
  u16*   aoT    = (u16*)  (ws + alloc((size_t)BSTOT * GA * EDIM * 2));        // 16 MB

  const size_t WSTRIDE = (size_t)GA * EDIM * EDIM;   // u16 elems per weight

  // 1) precision / layout prep
  prep_x_kernel<<<32768, 256, 0, stream>>>(x, xbT);
  prep_w_kernel<<<32768, 256, 0, stream>>>(wq, wk, wv, wo, wbT);

  // 2) q/k/v geo-linears + versor norm + attention packing (reuse gbuf)
  dim3 geoGrid(NGEO / 128, EDIM / 128, 1);
  geo_gemm_kernel<<<geoGrid, 256, 0, stream>>>(wbT + 0 * WSTRIDE, xbT, gbuf);
  norm_pack_kernel<0><<<BSTOT, 256, 0, stream>>>(gbuf, qsig, qn2);
  geo_gemm_kernel<<<geoGrid, 256, 0, stream>>>(wbT + 1 * WSTRIDE, xbT, gbuf);
  norm_pack_kernel<1><<<BSTOT, 256, 0, stream>>>(gbuf, kbf, kn2);
  geo_gemm_kernel<<<geoGrid, 256, 0, stream>>>(wbT + 2 * WSTRIDE, xbT, gbuf);
  norm_pack_kernel<2><<<BSTOT, 256, 0, stream>>>(gbuf, vT, nullptr);

  // 3) scalar scores: qsig . k  (M=512,N=512,K=1024, batch 16)
  gemm_kernel<0, 1024, 512><<<dim3(4, 4, BHCNT), 256, 0, stream>>>(qsig, kbf, sf32, nullptr, nullptr);

  // 4) bivector term + scale; emit score^T bf16 for triangle GEMM
  score_fix_kernel<<<(BHCNT * SEQ * SEQ) / 256, 256, 0, stream>>>(sf32, qn2, kn2, gamma, scoreT);

  // 5) triangle update: P = softmax(score); score = score + 0.1 * P@score; mask
  softmax_kernel<<<BHCNT * SEQ, 256, 0, stream>>>(sf32, Pbuf);
  gemm_kernel<1, 512, 512><<<dim3(4, 4, BHCNT), 256, 0, stream>>>(Pbuf, scoreT, sf32, nullptr, maskp);

  // 6) attn = softmax(score); out = attn @ v  -> aoT (geo-linear input layout)
  softmax_kernel<<<BHCNT * SEQ, 256, 0, stream>>>(sf32, Pbuf);
  gemm_kernel<2, 512, 1024><<<dim3(8, 4, BHCNT), 256, 0, stream>>>(Pbuf, vT, nullptr, aoT, nullptr);

  // 7) output geo-linear + versor norm, straight into d_out
  geo_gemm_kernel<<<geoGrid, 256, 0, stream>>>(wbT + 3 * WSTRIDE, aoT, outp);
  final_norm_kernel<<<BSTOT, 256, 0, stream>>>(outp);
}
